// DeltaRNNGLM_5042291605670
// MI455X (gfx1250) — compile-verified, matrix operand order checked by reading, not run on hardware
//
#include <hip/hip_runtime.h>
#include <math.h>

typedef _Float16 h16_t;
typedef __attribute__((ext_vector_type(16))) _Float16 v16h;
typedef __attribute__((ext_vector_type(8)))  _Float16 v8h;
typedef __attribute__((ext_vector_type(8)))  float    v8f;

#define BFULL 256
#define TLEN  4096
#define FDIM  3
#define HDIM  128
#define SDIM  2
#define G3H   384          // 3*H
#define BB    16           // batch tile per block
#define NTHR  256          // 8 waves of 32

// ---------------------------------------------------------------------------
// Fragment loaders.  16-bit A-matrix 16x32 per-lane layout (ISA 7.12.2):
//  lanes 0-15 : halfs 0-7 -> K=0..7,  halfs 8-15 -> K=16..23
//  lanes 16-31: halfs 0-7 -> K=8..15, halfs 8-15 -> K=24..31
// i.e. two contiguous 8-half (16B) runs per lane.  We use the mirrored
// mapping for B (lane%16 selects N column, same K runs along the weight row).
// ---------------------------------------------------------------------------
__device__ __forceinline__ v16h load_a_frag_lds(const h16_t* base, int lane, int kc) {
    int m = lane & 15, g = lane >> 4;
    const h16_t* p1 = base + m * HDIM + kc * 32 + g * 8;       // 16B aligned
    v8h lo = *(const v8h*)p1;
    v8h hi = *(const v8h*)(p1 + 16);
    v16h r;
#pragma unroll
    for (int j = 0; j < 8; ++j) { r[j] = lo[j]; r[j + 8] = hi[j]; }
    return r;
}

__device__ __forceinline__ v16h load_b_frag_glb(const h16_t* w, int ktot, int N0,
                                                int lane, int kc) {
    int n = lane & 15, g = lane >> 4;
    const h16_t* p1 = w + (size_t)(N0 + n) * ktot + kc * 32 + g * 8; // 16B aligned
    v8h lo = *(const v8h*)p1;
    v8h hi = *(const v8h*)(p1 + 16);
    v16h r;
#pragma unroll
    for (int j = 0; j < 8; ++j) { r[j] = lo[j]; r[j + 8] = hi[j]; }
    return r;
}

__device__ __forceinline__ float sigmoidf_(float v) { return 1.0f / (1.0f + __expf(-v)); }

// ---------------------------------------------------------------------------
// One-time weight conversion: f32 -> f16 into workspace.
//   d_ws[0        .. 16383] : dnn_w1  (128x128), row-major [out][in]
//   d_ws[16384 .. 65535]    : gru_w_hh (384x128), row-major [out][in]
// ---------------------------------------------------------------------------
__global__ void cvt_weights(const float* __restrict__ w1,
                            const float* __restrict__ whh,
                            h16_t* __restrict__ dst) {
    int i = blockIdx.x * blockDim.x + threadIdx.x;
    if (i < HDIM * HDIM)       dst[i] = (h16_t)w1[i];
    if (i < G3H * HDIM)        dst[HDIM * HDIM + i] = (h16_t)whh[i];
}

// ---------------------------------------------------------------------------
// Main persistent kernel: one block handles 16 batch elements for all 4096
// steps.  Weights live in registers (WMMA B-fragments) across the time loop.
// ---------------------------------------------------------------------------
__global__ __launch_bounds__(NTHR) void delta_rnn_kernel(
    const float* __restrict__ x,      const float* __restrict__ W0,
    const float* __restrict__ h0,
    const float* __restrict__ w0,     const float* __restrict__ b0,
    const float* __restrict__ b1,
    const float* __restrict__ w2,     const float* __restrict__ b2,
    const float* __restrict__ wih,    const float* __restrict__ bih,
    const float* __restrict__ bhh,
    const float* __restrict__ fcw,    const float* __restrict__ scal,
    const h16_t* __restrict__ w1h,    const h16_t* __restrict__ whhh,
    float* __restrict__ out_sig,      float* __restrict__ out_dw,
    float* __restrict__ out_W) {

    __shared__ float  g[BB * G3H];        // gi(+b_hh) then gi+gh   (24 KB)
    __shared__ float  gin[BB * HDIM];     // gi for n-gate           (8 KB)
    __shared__ h16_t  a0s[BB * HDIM];     // dnn layer0 act (f16)
    __shared__ h16_t  a1s[BB * HDIM];     // dnn layer1 act (f16)
    __shared__ h16_t  hs[BB * HDIM];      // hidden state  (f16)
    __shared__ float  w0s[HDIM * 4];
    __shared__ float  b0s[HDIM], b1s[HDIM];
    __shared__ float  wihs[G3H * 2];
    __shared__ float  bihs[G3H], bhhs[G3H];
    __shared__ float  w2s[SDIM * HDIM], fcws[SDIM * HDIM];
    __shared__ float  b2s[SDIM], scals[SDIM];
    __shared__ float  xts[BB * SDIM];     // dnn output = GRU input
    __shared__ float  Wst[BB * SDIM];     // integrated weight state

    const int tid  = threadIdx.x;
    const int lane = tid & 31;
    const int wv   = tid >> 5;            // wave id 0..7
    const int bg0  = blockIdx.x * BB;

    // ---- preload constants / state into LDS --------------------------------
    for (int i = tid; i < HDIM * 4; i += NTHR) w0s[i] = w0[i];
    for (int i = tid; i < HDIM; i += NTHR) { b0s[i] = b0[i]; b1s[i] = b1[i]; }
    for (int i = tid; i < G3H * 2; i += NTHR) wihs[i] = wih[i];
    for (int i = tid; i < G3H; i += NTHR) { bihs[i] = bih[i]; bhhs[i] = bhh[i]; }
    for (int i = tid; i < SDIM * HDIM; i += NTHR) { w2s[i] = w2[i]; fcws[i] = fcw[i]; }
    if (tid < SDIM) { b2s[tid] = b2[tid]; scals[tid] = scal[tid]; }
    for (int i = tid; i < BB * SDIM; i += NTHR)
        Wst[i] = W0[(size_t)bg0 * SDIM + i];
    for (int i = tid; i < BB * HDIM; i += NTHR)
        hs[i] = (h16_t)h0[(size_t)bg0 * HDIM + i];

    // ---- load WMMA B-fragments into registers (resident for the t-loop) ---
    v16h w1frag[4];
#pragma unroll
    for (int kc = 0; kc < 4; ++kc)
        w1frag[kc] = load_b_frag_glb(w1h, HDIM, 16 * wv, lane, kc);
    v16h whhfrag[3][4];
#pragma unroll
    for (int c = 0; c < 3; ++c) {
        int N0 = 16 * (wv + 8 * c);
#pragma unroll
        for (int kc = 0; kc < 4; ++kc)
            whhfrag[c][kc] = load_b_frag_glb(whhh, HDIM, N0, lane, kc);
    }
    __syncthreads();

    const int nD = lane & 15;             // D/C layout: column
    const int mb = (lane >> 4) * 8;       // D/C layout: row base

    for (int t = 0; t < TLEN; ++t) {
        // ---- A: dnn layer 0 (4 -> 128), VALU, write f16 to LDS -----------
        {
            int b = tid >> 4;
            size_t xb = ((size_t)(bg0 + b) * TLEN + t) * FDIM;
            float x1v = x[xb + 1], x2v = x[xb + 2];
            float Wv0 = Wst[b * 2 + 0], Wv1 = Wst[b * 2 + 1];
#pragma unroll
            for (int k = 0; k < 8; ++k) {
                int j = (tid & 15) * 8 + k;
                float a = Wv0 * w0s[j * 4 + 0] + Wv1 * w0s[j * 4 + 1] +
                          x1v * w0s[j * 4 + 2] + x2v * w0s[j * 4 + 3] + b0s[j];
                a0s[b * HDIM + j] = (h16_t)(a > 0.0f ? a : 0.0f);
            }
        }
        __syncthreads();

        // ---- B: dnn layer 1 (128x128) via WMMA, relu, f16 to LDS ---------
        {
            int N0 = 16 * wv;
            float bv = b1s[N0 + nD];
            v8f acc;
#pragma unroll
            for (int r = 0; r < 8; ++r) acc[r] = bv;
#pragma unroll
            for (int kc = 0; kc < 4; ++kc) {
                v16h a = load_a_frag_lds(a0s, lane, kc);
                acc = __builtin_amdgcn_wmma_f32_16x16x32_f16(
                    false, a, false, w1frag[kc], (short)0, acc, false, false);
            }
#pragma unroll
            for (int r = 0; r < 8; ++r) {
                float v = acc[r];
                a1s[(mb + r) * HDIM + N0 + nD] = (h16_t)(v > 0.0f ? v : 0.0f);
            }
        }
        __syncthreads();

        // ---- C1: dnn layer 2 (128 -> 2) -> GRU input xt ------------------
        if (tid < BB * SDIM) {
            int b = tid >> 1, s = tid & 1;
            float acc = b2s[s];
#pragma unroll 8
            for (int k = 0; k < HDIM; ++k)
                acc += (float)a1s[b * HDIM + k] * w2s[s * HDIM + k];
            xts[b * 2 + s] = acc;
        }
        __syncthreads();

        // ---- C2: gi = xt @ w_ih^T + b_ih; stage biases for WMMA-C --------
        {
            int b = tid >> 4;
            float x0v = xts[b * 2 + 0], x1v = xts[b * 2 + 1];
#pragma unroll
            for (int k = 0; k < 24; ++k) {
                int j = (tid & 15) + (k << 4);
                float giv = x0v * wihs[j * 2 + 0] + x1v * wihs[j * 2 + 1] + bihs[j];
                if (j < 2 * HDIM) {
                    g[b * G3H + j] = giv + bhhs[j];      // r,z: gi + b_hh
                } else {
                    g[b * G3H + j] = bhhs[j];            // n: accumulate gh_n here
                    gin[b * HDIM + (j - 2 * HDIM)] = giv; // keep gi_n separate
                }
            }
        }
        __syncthreads();

        // ---- D: gh = h @ w_hh^T via WMMA, C initialized from LDS ---------
        {
            v16h afr[4];
#pragma unroll
            for (int kc = 0; kc < 4; ++kc)
                afr[kc] = load_a_frag_lds(hs, lane, kc);
#pragma unroll
            for (int c = 0; c < 3; ++c) {
                int N0 = 16 * (wv + 8 * c);
                v8f acc;
#pragma unroll
                for (int r = 0; r < 8; ++r) acc[r] = g[(mb + r) * G3H + N0 + nD];
#pragma unroll
                for (int kc = 0; kc < 4; ++kc)
                    acc = __builtin_amdgcn_wmma_f32_16x16x32_f16(
                        false, afr[kc], false, whhfrag[c][kc], (short)0, acc,
                        false, false);
#pragma unroll
                for (int r = 0; r < 8; ++r) g[(mb + r) * G3H + N0 + nD] = acc[r];
            }
        }
        __syncthreads();

        // ---- E: GRU gates, new hidden state ------------------------------
        {
            int b = tid >> 5;                       // 2 rows per wave... no:
            // 2048 elems / 256 threads = 8 each, contiguous per batch row
            int bb = tid >> 4;
#pragma unroll
            for (int k = 0; k < 8; ++k) {
                int j = (tid & 15) * 8 + k;
                float rg  = sigmoidf_(g[bb * G3H + j]);
                float zg  = sigmoidf_(g[bb * G3H + HDIM + j]);
                float ghn = g[bb * G3H + 2 * HDIM + j];
                float inn = gin[bb * HDIM + j];
                float ng  = tanhf(inn + rg * ghn);
                float hov = (float)hs[bb * HDIM + j];
                hs[bb * HDIM + j] = (h16_t)((1.0f - zg) * ng + zg * hov);
            }
            (void)b;
        }
        __syncthreads();

        // ---- F1: dW = (h @ fc_w^T)*scaling; W += dW; emit outputs --------
        if (tid < BB * SDIM) {
            int b = tid >> 1, s = tid & 1;
            float acc = 0.0f;
#pragma unroll 8
            for (int k = 0; k < HDIM; ++k)
                acc += (float)hs[b * HDIM + k] * fcws[s * HDIM + k];
            float dW = acc * scals[s];
            float Wv = Wst[b * 2 + s] + dW;
            Wst[b * 2 + s] = Wv;
            size_t og = ((size_t)(bg0 + b) * TLEN + t) * SDIM + s;
            out_W[og] = Wv;
            if (t == TLEN - 1) out_dw[(size_t)(bg0 + b) * SDIM + s] = dW;
        }
        __syncthreads();

        // ---- F2: sigmoid head -------------------------------------------
        if (tid < BB) {
            int b = tid;
            float wv0 = Wst[b * 2 + 0], wv1 = Wst[b * 2 + 1];
            size_t xb = ((size_t)(bg0 + b) * TLEN + t) * FDIM;
            float wt = wv0 * x[xb] + wv1;
            out_sig[(size_t)(bg0 + b) * TLEN + t] = sigmoidf_(wt);
        }
        __syncthreads();
    }
}

// ---------------------------------------------------------------------------
extern "C" void kernel_launch(void* const* d_in, const int* in_sizes, int n_in,
                              void* d_out, int out_size, void* d_ws, size_t ws_size,
                              hipStream_t stream) {
    const float* x    = (const float*)d_in[0];
    const float* W0   = (const float*)d_in[1];
    const float* h0   = (const float*)d_in[2];
    const float* w0   = (const float*)d_in[3];
    const float* b0   = (const float*)d_in[4];
    const float* w1   = (const float*)d_in[5];
    const float* b1   = (const float*)d_in[6];
    const float* w2   = (const float*)d_in[7];
    const float* b2   = (const float*)d_in[8];
    const float* wih  = (const float*)d_in[9];
    const float* whh  = (const float*)d_in[10];
    const float* bih  = (const float*)d_in[11];
    const float* bhh  = (const float*)d_in[12];
    const float* fcw  = (const float*)d_in[13];
    const float* scal = (const float*)d_in[14];

    h16_t* wgt16 = (h16_t*)d_ws;                 // 65536 halfs = 128 KB
    const h16_t* w1h  = wgt16;
    const h16_t* whhh = wgt16 + HDIM * HDIM;

    float* out_sig = (float*)d_out;                          // B*T
    float* out_dw  = out_sig + (size_t)BFULL * TLEN;         // B*S
    float* out_W   = out_dw + (size_t)BFULL * SDIM;          // B*T*S

    cvt_weights<<<(G3H * HDIM + 255) / 256, 256, 0, stream>>>(w1, whh, wgt16);

    delta_rnn_kernel<<<BFULL / BB, NTHR, 0, stream>>>(
        x, W0, h0, w0, b0, b1, w2, b2, wih, bih, bhh, fcw, scal,
        w1h, whhh, out_sig, out_dw, out_W);
}
